// GPSA_23467701305477
// MI455X (gfx1250) — matrix operand order repaired and checked by reading, not compiled, not run on hardware
//
#include <hip/hip_runtime.h>
#include <hip/hip_bf16.h>

// ---------------------------------------------------------------------------
// GPSA (gated positional self-attention) for MI455X / gfx1250, wave32 + WMMA.
// B=16, N=576 (24x24 grid), C=768, H=12, hd=64, scale=1/8.
// Strategy: f16 WMMA (f32 accumulate) for all GEMMs; attention fully fused per
// (b, h, 16-row block) so the [B,H,N,N] attention tensor never touches HBM.
// GEMMs use double-buffered LDS staging (one barrier per K-step) and
// global_prefetch for the K-tile two steps ahead.
// ---------------------------------------------------------------------------

typedef __attribute__((ext_vector_type(16))) _Float16 v16h;
typedef __attribute__((ext_vector_type(8)))  _Float16 v8h;
typedef __attribute__((ext_vector_type(8)))  float    v8f;

#define BATCH 16
#define NTOK  576
#define CDIM  768
#define NHEAD 12
#define HD    64
#define MTOT  (BATCH * NTOK)      // 9216
#define NQKV  (3 * CDIM)          // 2304 fused output columns (q|k|v)

// ---- WMMA fragment helpers (CDNA5 16x16x32 f16 layouts, wave32) -----------
// A-matrix 16x32: lane L -> row L%16; K elements {kt..kt+7, kt+16..kt+23},
// kt = (L>=16)?8:0. rowp points at (row, K=0); K contiguous in memory.
__device__ inline v16h fragA(const _Float16* rowp, int lane) {
    const int kt = (lane & 16) >> 1;            // 0 or 8 halves
    v8h lo = *(const v8h*)(rowp + kt);
    v8h hi = *(const v8h*)(rowp + kt + 16);
    return __builtin_shufflevector(lo, hi, 0,1,2,3,4,5,6,7,8,9,10,11,12,13,14,15);
}
// B-matrix 32x16: lane L -> col L%16; K elements {kt..kt+7, kt+16..kt+23},
// kt = (L>=16)?8:0. colp points at (K=0, col); K contiguous in memory.
__device__ inline v16h fragB(const _Float16* colp, int lane) {
    const int kt = (lane & 16) >> 1;            // 0 or 8 halves
    v8h lo = *(const v8h*)(colp + kt);
    v8h hi = *(const v8h*)(colp + kt + 16);
    return __builtin_shufflevector(lo, hi, 0,1,2,3,4,5,6,7,8,9,10,11,12,13,14,15);
}
__device__ inline v8f wmma16(v16h a, v16h b, v8f c) {
    return __builtin_amdgcn_wmma_f32_16x16x32_f16(false, a, false, b, (short)0, c,
                                                  false, false);
}

// ---------------------------------------------------------------------------
// Kernel 1: positional softmax  pos[h][n][m] (f16), batch-invariant.
// logits = Wpos[h]·(dx, dy, dx^2+dy^2) + bpos[h]; softmax over m.
// 256 threads = 16 rows x 16 col-lanes, 36 cols per lane.
// ---------------------------------------------------------------------------
__global__ __launch_bounds__(256) void gpsa_pos_kernel(
    const float* __restrict__ Wpos, const float* __restrict__ bpos,
    _Float16* __restrict__ pos)
{
    const int blk = blockIdx.x;                 // h*36 + nt
    const int nt  = blk % 36;
    const int h   = blk / 36;
    const int tid = threadIdx.x;
    const int r   = tid >> 4;
    const int c0  = tid & 15;
    const int n   = nt * 16 + r;
    const int nx  = n % 24, ny = n / 24;
    const float w0 = Wpos[h*3+0], w1 = Wpos[h*3+1], w2 = Wpos[h*3+2], bb = bpos[h];

    float e[36];
    float mx = -1e30f;
    #pragma unroll
    for (int i = 0; i < 36; ++i) {
        const int m = c0 + i * 16;
        const float dx = (float)(m % 24 - nx);
        const float dy = (float)(m / 24 - ny);
        const float lg = w0 * dx + w1 * dy + w2 * (dx*dx + dy*dy) + bb;
        e[i] = lg;
        mx = fmaxf(mx, lg);
    }
    #pragma unroll
    for (int o = 8; o >= 1; o >>= 1) mx = fmaxf(mx, __shfl_xor(mx, o, 16));
    float sum = 0.f;
    #pragma unroll
    for (int i = 0; i < 36; ++i) { e[i] = __expf(e[i] - mx); sum += e[i]; }
    #pragma unroll
    for (int o = 8; o >= 1; o >>= 1) sum += __shfl_xor(sum, o, 16);
    const float inv = 1.f / sum;
    _Float16* prow = pos + ((size_t)h * NTOK + n) * NTOK;
    #pragma unroll
    for (int i = 0; i < 36; ++i) prow[c0 + i * 16] = (_Float16)(e[i] * inv);
}

// ---------------------------------------------------------------------------
// Kernel 2: fused QKV projection GEMM.  [9216 x 768] x [768 x 2304].
// Bmat[k][j] = Wcat[j][k], Wcat = rows(Wqk) ++ rows(Wv) (row-major [j][k], so
// B tiles are contiguous-in-k loads). 128x64 block tile, BK=32, 8 waves in a
// 4x2 grid, each wave owns a 32x32 micro-tile = 4 WMMAs per K-step.
// Double-buffered LDS: stage tile t+1 while computing tile t; 1 barrier/step.
// Epilogue scatters to per-head layouts: q (x scale) & k as [B,H,N,hd],
// v transposed to [B,H,hd,N] so attn@v B-fragments are contiguous.
// ---------------------------------------------------------------------------
#define BM 128
#define BN 64
#define BK 32
#define LDS_K 40    // padded halves per row (80B, 16B-aligned rows)
#define NKSTEP (CDIM / BK)   // 24

__global__ __launch_bounds__(256) void gpsa_qkv_gemm(
    const float* __restrict__ x, const float* __restrict__ Wqk,
    const float* __restrict__ Wv,
    _Float16* __restrict__ q, _Float16* __restrict__ k,
    _Float16* __restrict__ vT)
{
    __shared__ alignas(16) _Float16 As[2][BM * LDS_K];
    __shared__ alignas(16) _Float16 Bs[2][BN * LDS_K];

    const int n0   = blockIdx.x * BN;
    const int m0   = blockIdx.y * BM;
    const int tid  = threadIdx.x;
    const int lane = tid & 31;
    const int wave = tid >> 5;
    const int wr   = wave >> 1;                 // 0..3
    const int wc   = wave & 1;                  // 0..1

    // per-thread staging geometry (fixed across K-steps)
    const int arow = tid >> 1, aseg = (tid & 1) * 16;
    const int bcol = tid >> 2, bseg = (tid & 3) * 8;
    const int j    = n0 + bcol;
    const float* asrc0 = x + (size_t)(m0 + arow) * CDIM + aseg;
    const float* bsrc0 = (j < 2 * CDIM)
        ? (Wqk + (size_t)j * CDIM + bseg)
        : (Wv  + (size_t)(j - 2 * CDIM) * CDIM + bseg);

    auto stage = [&](int kt, int buf) {
        const float* asrc = asrc0 + kt * BK;
        _Float16* adst = As[buf] + arow * LDS_K + aseg;
        #pragma unroll
        for (int i = 0; i < 16; ++i) adst[i] = (_Float16)asrc[i];
        const float* bsrc = bsrc0 + kt * BK;
        _Float16* bdst = Bs[buf] + bcol * LDS_K + bseg;
        #pragma unroll
        for (int i = 0; i < 8; ++i) bdst[i] = (_Float16)bsrc[i];
        // pull the tile two K-steps ahead toward the caches (gfx1250 prefetch)
        if (kt + 2 < NKSTEP) {
            __builtin_prefetch(asrc + 2 * BK, 0, 1);
            __builtin_prefetch(bsrc + 2 * BK, 0, 1);
        }
    };

    v8f acc[2][2] = {};
    stage(0, 0);
    __syncthreads();

    for (int kt = 0; kt < NKSTEP; ++kt) {
        const int cur = kt & 1;
        if (kt + 1 < NKSTEP) stage(kt + 1, cur ^ 1);   // overlap with WMMAs
        #pragma unroll
        for (int tc = 0; tc < 2; ++tc) {
            v16h bf = fragB(Bs[cur] + (wc*32 + tc*16 + (lane & 15)) * LDS_K, lane);
            #pragma unroll
            for (int tr = 0; tr < 2; ++tr) {
                v16h af = fragA(As[cur] + (wr*32 + tr*16 + (lane & 15)) * LDS_K, lane);
                acc[tr][tc] = wmma16(af, bf, acc[tr][tc]);
            }
        }
        __syncthreads();
    }

    const float scale = 0.125f;                 // hd^-0.5, folded into q
    #pragma unroll
    for (int tr = 0; tr < 2; ++tr)
    #pragma unroll
    for (int tc = 0; tc < 2; ++tc)
    #pragma unroll
    for (int e = 0; e < 8; ++e) {
        const int m = m0 + wr*32 + tr*16 + e + ((lane & 16) >> 1);
        const int jj = n0 + wc*32 + tc*16 + (lane & 15);
        const int b = m / NTOK, n = m % NTOK;
        const float val = acc[tr][tc][e];
        if (jj < CDIM) {                        // q: reshape(B,N,2,H,hd) slot 0
            const int hh = jj >> 6, d = jj & 63;
            q[(((size_t)b*NHEAD + hh)*NTOK + n)*HD + d] = (_Float16)(val * scale);
        } else if (jj < 2 * CDIM) {             // k: slot 1
            const int j2 = jj - CDIM, hh = j2 >> 6, d = j2 & 63;
            k[(((size_t)b*NHEAD + hh)*NTOK + n)*HD + d] = (_Float16)val;
        } else {                                // v: transposed [B,H,hd,N]
            const int j2 = jj - 2*CDIM, hh = j2 >> 6, d = j2 & 63;
            vT[(((size_t)b*NHEAD + hh)*HD + d)*NTOK + n] = (_Float16)val;
        }
    }
}

// ---------------------------------------------------------------------------
// Kernel 3: fused attention for one (b, h, 16-row block).
// Phase 1: scores S = (q*scale)·k^T via WMMA (8 waves split 36 col-tiles).
// Phase 2: row softmax, sigmoid-gated blend with pos_score, renormalize.
// Phase 3: ctx = attn @ v via WMMA (4 col-tiles x 2 K-halves across 8 waves).
// The 255 MB attention tensor never leaves LDS.
// ---------------------------------------------------------------------------
#define SLD 592     // padded row stride (floats / halves), 16B-aligned rows

__global__ __launch_bounds__(256) void gpsa_attn_kernel(
    const _Float16* __restrict__ q, const _Float16* __restrict__ kmat,
    const _Float16* __restrict__ vT, const _Float16* __restrict__ pos,
    const float* __restrict__ gating, _Float16* __restrict__ ctx)
{
    __shared__ alignas(16) float    S[16 * SLD];       // scores / probs (f32)
    __shared__ alignas(16) _Float16 A16[16 * SLD];     // blended attn (f16)
    __shared__ alignas(16) float    OutP[2 * 16 * HD]; // K-half partials

    const int blk = blockIdx.x;
    const int nt  = blk % 36;
    const int bh  = blk / 36;                   // b*NHEAD + h
    const int h   = bh % NHEAD;
    const int b   = bh / NHEAD;
    const int n0  = nt * 16;
    const int tid = threadIdx.x, lane = tid & 31, wave = tid >> 5;

    const _Float16* qbase = q    + (size_t)bh * NTOK * HD;
    const _Float16* kbase = kmat + (size_t)bh * NTOK * HD;

    // q fragments for the block's 16 rows (d contiguous in memory)
    const v16h qa0 = fragA(qbase + (size_t)(n0 + (lane & 15)) * HD +  0, lane);
    const v16h qa1 = fragA(qbase + (size_t)(n0 + (lane & 15)) * HD + 32, lane);

    for (int mt = wave; mt < 36; mt += 8) {
        const int mcol = mt * 16 + (lane & 15);
        if (mt + 8 < 36)                        // prefetch next k-tile row
            __builtin_prefetch(kbase + (size_t)(mcol + 128) * HD, 0, 1);
        const v16h b0 = fragB(kbase + (size_t)mcol * HD +  0, lane);
        const v16h b1 = fragB(kbase + (size_t)mcol * HD + 32, lane);
        v8f acc = {};
        acc = wmma16(qa0, b0, acc);
        acc = wmma16(qa1, b1, acc);
        const int row0 = (lane & 16) >> 1;
        #pragma unroll
        for (int e = 0; e < 8; ++e)
            S[(row0 + e) * SLD + mt * 16 + (lane & 15)] = acc[e];
    }
    __syncthreads();

    // ---- softmax + gated blend (16 lanes per row) ----
    {
        const int r = tid >> 4, c0 = tid & 15;
        float* Srow = S + r * SLD;
        float mx = -1e30f;
        for (int m = c0; m < NTOK; m += 16) mx = fmaxf(mx, Srow[m]);
        #pragma unroll
        for (int o = 8; o >= 1; o >>= 1) mx = fmaxf(mx, __shfl_xor(mx, o, 16));
        float sum = 0.f;
        for (int m = c0; m < NTOK; m += 16) {
            const float e = __expf(Srow[m] - mx);
            Srow[m] = e; sum += e;
        }
        #pragma unroll
        for (int o = 8; o >= 1; o >>= 1) sum += __shfl_xor(sum, o, 16);
        const float g    = 1.f / (1.f + __expf(-gating[h]));
        const float invc = (1.f - g) / sum;
        const _Float16* prow = pos + ((size_t)h * NTOK + n0 + r) * NTOK;
        float sum2 = 0.f;
        for (int m = c0; m < NTOK; m += 16) {
            const float a = Srow[m] * invc + g * (float)prow[m];
            Srow[m] = a; sum2 += a;
        }
        #pragma unroll
        for (int o = 8; o >= 1; o >>= 1) sum2 += __shfl_xor(sum2, o, 16);
        const float inv2 = 1.f / sum2;
        _Float16* arow = A16 + r * SLD;
        for (int m = c0; m < NTOK; m += 16)
            arow[m] = (_Float16)(Srow[m] * inv2);
    }
    __syncthreads();

    // ---- attn @ v: wave -> (col tile ct, K half kh) ----
    {
        const int ct = wave & 3, kh = wave >> 2;
        const _Float16* vbase =
            vT + ((size_t)bh * HD + ct * 16 + (lane & 15)) * NTOK; // m contiguous
        v8f acc = {};
        for (int s = 0; s < 9; ++s) {
            const int kk0 = (kh * 9 + s) * 32;
            if (s + 1 < 9)
                __builtin_prefetch(vbase + kk0 + 32, 0, 1);
            const v16h af = fragA(A16 + (size_t)(lane & 15) * SLD + kk0, lane);
            const v16h bf = fragB(vbase + kk0, lane);
            acc = wmma16(af, bf, acc);
        }
        const int row0 = (lane & 16) >> 1;
        float* op = OutP + kh * 16 * HD;
        #pragma unroll
        for (int e = 0; e < 8; ++e)
            op[(row0 + e) * HD + ct * 16 + (lane & 15)] = acc[e];
    }
    __syncthreads();

    // combine K-halves, emit ctx[(b*N + n)*C + h*hd + d] as f16
    for (int i = tid; i < 16 * HD; i += 256) {
        const int rr = i >> 6, d = i & 63;
        const float o = OutP[i] + OutP[16 * HD + i];
        ctx[((size_t)(b * NTOK + n0 + rr)) * CDIM + h * HD + d] = (_Float16)o;
    }
}

// ---------------------------------------------------------------------------
// Kernel 4: output projection.  out = ctx(f16) @ Wproj^T + bproj  (f32 out).
// Same double-buffered 128x64 WMMA tiling as kernel 2.
// ---------------------------------------------------------------------------
__global__ __launch_bounds__(256) void gpsa_proj_gemm(
    const _Float16* __restrict__ ctx, const float* __restrict__ Wproj,
    const float* __restrict__ bproj, float* __restrict__ out)
{
    __shared__ alignas(16) _Float16 As[2][BM * LDS_K];
    __shared__ alignas(16) _Float16 Bs[2][BN * LDS_K];

    const int n0   = blockIdx.x * BN;
    const int m0   = blockIdx.y * BM;
    const int tid  = threadIdx.x;
    const int lane = tid & 31;
    const int wave = tid >> 5;
    const int wr   = wave >> 1;
    const int wc   = wave & 1;

    const int arow = tid >> 1, aseg = (tid & 1) * 16;
    const int bcol = tid >> 2, bseg = (tid & 3) * 8;
    const _Float16* asrc0 = ctx + (size_t)(m0 + arow) * CDIM + aseg;
    const float*    bsrc0 = Wproj + (size_t)(n0 + bcol) * CDIM + bseg;

    auto stage = [&](int kt, int buf) {
        const _Float16* asrc = asrc0 + kt * BK;       // already f16
        _Float16* adst = As[buf] + arow * LDS_K + aseg;
        *(v8h*)(adst)     = *(const v8h*)(asrc);
        *(v8h*)(adst + 8) = *(const v8h*)(asrc + 8);
        const float* bsrc = bsrc0 + kt * BK;
        _Float16* bdst = Bs[buf] + bcol * LDS_K + bseg;
        #pragma unroll
        for (int i = 0; i < 8; ++i) bdst[i] = (_Float16)bsrc[i];
        if (kt + 2 < NKSTEP) {
            __builtin_prefetch(asrc + 2 * BK, 0, 1);
            __builtin_prefetch(bsrc + 2 * BK, 0, 1);
        }
    };

    v8f acc[2][2] = {};
    stage(0, 0);
    __syncthreads();

    for (int kt = 0; kt < NKSTEP; ++kt) {
        const int cur = kt & 1;
        if (kt + 1 < NKSTEP) stage(kt + 1, cur ^ 1);
        #pragma unroll
        for (int tc = 0; tc < 2; ++tc) {
            v16h bf = fragB(Bs[cur] + (wc*32 + tc*16 + (lane & 15)) * LDS_K, lane);
            #pragma unroll
            for (int tr = 0; tr < 2; ++tr) {
                v16h af = fragA(As[cur] + (wr*32 + tr*16 + (lane & 15)) * LDS_K, lane);
                acc[tr][tc] = wmma16(af, bf, acc[tr][tc]);
            }
        }
        __syncthreads();
    }

    #pragma unroll
    for (int tr = 0; tr < 2; ++tr)
    #pragma unroll
    for (int tc = 0; tc < 2; ++tc)
    #pragma unroll
    for (int e = 0; e < 8; ++e) {
        const int m = m0 + wr*32 + tr*16 + e + ((lane & 16) >> 1);
        const int jj = n0 + wc*32 + tc*16 + (lane & 15);
        out[(size_t)m * CDIM + jj] = acc[tr][tc][e] + bproj[jj];
    }
}

// ---------------------------------------------------------------------------
extern "C" void kernel_launch(void* const* d_in, const int* in_sizes, int n_in,
                              void* d_out, int out_size, void* d_ws, size_t ws_size,
                              hipStream_t stream)
{
    (void)in_sizes; (void)n_in; (void)out_size;
    const float* x      = (const float*)d_in[0];
    const float* Wqk    = (const float*)d_in[1];
    const float* Wv     = (const float*)d_in[2];
    const float* Wproj  = (const float*)d_in[3];
    const float* bproj  = (const float*)d_in[4];
    const float* Wpos   = (const float*)d_in[5];
    const float* bpos   = (const float*)d_in[6];
    const float* gating = (const float*)d_in[7];
    float* out = (float*)d_out;

    // ---- workspace carve-up (f16 intermediates, ~64.6 MB total) ----
    char* ws = (char*)d_ws;
    const size_t qkvBytes = (size_t)BATCH * NHEAD * NTOK * HD * sizeof(_Float16);
    const size_t posBytes = (size_t)NHEAD * NTOK * NTOK * sizeof(_Float16);
    const size_t ctxBytes = (size_t)MTOT * CDIM * sizeof(_Float16);
    if (ws_size < 3 * qkvBytes + posBytes + ctxBytes) return;

    _Float16* q   = (_Float16*)ws;             ws += qkvBytes;
    _Float16* k   = (_Float16*)ws;             ws += qkvBytes;
    _Float16* vT  = (_Float16*)ws;             ws += qkvBytes;
    _Float16* pos = (_Float16*)ws;             ws += posBytes;
    _Float16* ctx = (_Float16*)ws;             ws += ctxBytes;

    // 1) batch-invariant positional softmax  [H,N,N] f16
    gpsa_pos_kernel<<<NHEAD * (NTOK / 16), 256, 0, stream>>>(Wpos, bpos, pos);
    // 2) fused QKV projection (WMMA GEMM, scatters to attention layouts)
    gpsa_qkv_gemm<<<dim3(NQKV / BN, MTOT / BM), 256, 0, stream>>>(
        x, Wqk, Wv, q, k, vT);
    // 3) fused attention (scores + softmax + gated blend + attn@v), all in LDS
    gpsa_attn_kernel<<<BATCH * NHEAD * (NTOK / 16), 256, 0, stream>>>(
        q, k, vT, pos, gating, ctx);
    // 4) output projection + bias -> f32
    gpsa_proj_gemm<<<dim3(CDIM / BN, MTOT / BM), 256, 0, stream>>>(
        ctx, Wproj, bproj, out);
}
